// NodeDenoiser_16200616641106
// MI455X (gfx1250) — matrix operand order repaired and checked by reading, not compiled, not run on hardware
//
#include <hip/hip_runtime.h>
#include <hip/hip_bf16.h>

#define N_NODES 50000
#define N_EDGES 800000
#define DH 128
#define ETILES 4
#define NTILES 4

typedef unsigned short u16;
typedef __attribute__((ext_vector_type(16))) __bf16 v16bf;
typedef __attribute__((ext_vector_type(8))) float v8f;
typedef unsigned int v4u __attribute__((ext_vector_type(4)));
typedef int v8i __attribute__((ext_vector_type(8)));
typedef int v4i __attribute__((ext_vector_type(4)));

union ABfrag { uint4 q[2]; v16bf v; };

#if defined(__gfx1250__) && __has_builtin(__builtin_amdgcn_tensor_load_to_lds) && \
    __has_builtin(__builtin_amdgcn_s_wait_tensorcnt)
#define USE_TDM 1
#else
#define USE_TDM 0
#endif

__device__ __forceinline__ u16 f2bf(float f) {
  unsigned u = __float_as_uint(f);
  unsigned r = ((u >> 16) & 1u) + 0x7FFFu;
  return (u16)((u + r) >> 16);
}
__device__ __forceinline__ float silu_f(float x) {
  return x / (1.0f + __expf(-x));
}

#if USE_TDM
// 1-D TDM copy: nelem bf16 elements, global -> LDS. Issue from one wave only.
__device__ __forceinline__ void tdm_load_1d(const void* gsrc, void* ldst, unsigned nelem) {
  unsigned long long ga = (unsigned long long)gsrc;
  unsigned la = (unsigned)(unsigned long long)ldst;  // low 32 bits = LDS offset
  v4u g0;
  g0[0] = 1u;                                         // count=1, user mode
  g0[1] = la;                                         // lds_addr
  g0[2] = (unsigned)(ga & 0xFFFFFFFFu);               // global_addr[31:0]
  g0[3] = (unsigned)((ga >> 32) & 0x1FFFFFFu) | (2u << 30);  // addr[56:32] | type=2
  v8i g1;
  g1[0] = (int)(1u << 16);                            // data_size=1 (2 bytes)
  g1[1] = (int)((nelem & 0xFFFFu) << 16);             // tensor_dim0[15:0] @ bits 63:48
  g1[2] = (int)((nelem >> 16) & 0xFFFFu);             // tensor_dim0[31:16]; tensor_dim1=0
  g1[3] = (int)((nelem & 0xFFFFu) << 16);             // tile_dim0 @ bits 127:112
  g1[4] = 0;                                          // tile_dim1=0 (1-D), tile_dim2=0
  g1[5] = (int)nelem;                                 // tensor_dim0_stride[31:0]
  g1[6] = 0;
  g1[7] = 0;
  v4i z4 = {};
#if defined(__clang_major__) && (__clang_major__ >= 23)
  v8i z8 = {};
  __builtin_amdgcn_tensor_load_to_lds(g0, g1, z4, z4, z8, 0);
#else
  __builtin_amdgcn_tensor_load_to_lds(g0, g1, z4, z4, 0);
#endif
}
#endif

// ---------------------------------------------------------------------------
// Shared WMMA GEMM stage: c[8] += A(16xK) x B(Kx128-slice)
// A row fixed per lane; B double-buffered so ds_loads overlap WMMA.
// ---------------------------------------------------------------------------
template <int KSTEPS, int BSTRIDE>
__device__ __forceinline__ void gemm_acc(v8f (&c)[8], const u16* __restrict__ aRowPtr,
                                         const u16* __restrict__ bMat, int l15, int lhi) {
  const uint4* ar = (const uint4*)aRowPtr;
  for (int k = 0; k < KSTEPS; ++k) {
    ABfrag a;
    int kb = k * 4 + lhi * 2;
    a.q[0] = ar[kb];
    a.q[1] = ar[kb + 1];
    int ko = k * 4 + lhi * 2;
    ABfrag bb[2];
    {
      const uint4* br = (const uint4*)(bMat + (size_t)l15 * BSTRIDE);
      bb[0].q[0] = br[ko]; bb[0].q[1] = br[ko + 1];
    }
#pragma unroll
    for (int n = 0; n < 8; ++n) {
      if (n < 7) {
        const uint4* br = (const uint4*)(bMat + (size_t)((n + 1) * 16 + l15) * BSTRIDE);
        bb[(n + 1) & 1].q[0] = br[ko];
        bb[(n + 1) & 1].q[1] = br[ko + 1];
      }
      c[n] = __builtin_amdgcn_wmma_f32_16x16x32_bf16(false, a.v, false, bb[n & 1].v,
                                                     (short)0, c[n], false, false);
    }
  }
}

__device__ __forceinline__ void zero_acc(v8f (&c)[8]) {
#pragma unroll
  for (int n = 0; n < 8; ++n)
#pragma unroll
    for (int i = 0; i < 8; ++i) c[n][i] = 0.0f;
}

// ---------------------------------------------------------------------------
// Small prep: t_emb + cond_emb -> g, then fold g into ni layer-1 bias.
// ---------------------------------------------------------------------------
__global__ void prep_small(const float* tptr, const float* cond,
                           const float* tmW1, const float* tmb1,
                           const float* tmW2, const float* tmb2,
                           const float* cmW1, const float* cmb1,
                           const float* cmW2, const float* cmb2,
                           const float* niW1, const float* nib1,
                           float* gbias) {
  __shared__ float s_in[256];
  __shared__ float s_h[128];
  __shared__ float s_g[128];
  int j = threadIdx.x;  // 0..127
  float tv = tptr[0];
  if (j < 64) {
    float fr = __expf(-logf(10000.0f) * (float)j / 63.0f);
    float a = tv * fr;
    s_in[j] = sinf(a);
    s_in[64 + j] = cosf(a);
  }
  __syncthreads();
  float acc = tmb1[j];
  for (int i = 0; i < 128; ++i) acc += s_in[i] * tmW1[i * 128 + j];
  s_h[j] = silu_f(acc);
  __syncthreads();
  float te = tmb2[j];
  for (int i = 0; i < 128; ++i) te += s_h[i] * tmW2[i * 128 + j];
  __syncthreads();
  acc = cmb1[j];
  for (int i = 0; i < 256; ++i) acc += cond[i] * cmW1[i * 128 + j];
  s_h[j] = silu_f(acc);
  __syncthreads();
  float ce = cmb2[j];
  for (int i = 0; i < 128; ++i) ce += s_h[i] * cmW2[i * 128 + j];
  s_g[j] = te + ce;
  __syncthreads();
  float gb = nib1[j];
  for (int i = 0; i < 128; ++i) gb += s_g[i] * niW1[(2 + i) * 128 + j];
  gbias[j] = gb;
}

// ---------------------------------------------------------------------------
// Weight transpose + f32->bf16: W(K,Nn) row-major  ->  out(Nn,Kpad) bf16
// ---------------------------------------------------------------------------
__global__ void wt_kernel(const float* __restrict__ W, u16* __restrict__ out,
                          int K, int Nn, int Kpad) {
  int idx = blockIdx.x * blockDim.x + threadIdx.x;
  int total = Nn * Kpad;
  if (idx >= total) return;
  int n = idx / Kpad, k = idx - n * Kpad;
  float v = (k < K) ? W[k * Nn + n] : 0.0f;
  out[idx] = f2bf(v);
}

// ---------------------------------------------------------------------------
// Per-edge RBF features (16 gaussians) in bf16
// ---------------------------------------------------------------------------
__global__ void rbf_kernel(const float* __restrict__ x, const int* __restrict__ ei,
                           u16* __restrict__ rbf) {
  int e = blockIdx.x * blockDim.x + threadIdx.x;
  if (e >= N_EDGES) return;
  int s = ei[e], d = ei[N_EDGES + e];
  float dx = x[2 * s] - x[2 * d];
  float dy = x[2 * s + 1] - x[2 * d + 1];
  float r = sqrtf(dx * dx + dy * dy + 1e-8f);
  const float step = 1.41421356237309f / 15.0f;   // R_MAX / (N_RBF-1)
  const float gamma = 56.25f;                     // 1/(2*step^2)
  __align__(16) u16 o[16];
#pragma unroll
  for (int i = 0; i < 16; ++i) {
    float diff = r - step * (float)i;
    o[i] = f2bf(__expf(-gamma * diff * diff));
  }
  uint4* dst = (uint4*)(rbf + (size_t)e * 16);
  dst[0] = *(const uint4*)&o[0];
  dst[1] = *(const uint4*)&o[8];
}

// ---------------------------------------------------------------------------
// Node init: h = silu(x*W1[0:2] + gbias) @ W2 + b2   (stage2 via WMMA)
// LDS: sW2[128][128] @0 (32768) | sM[64][128] @32768 (16384)
//      sw0 @49152 | sw1 @49664 | sgb @50176  -> 50688 B
// ---------------------------------------------------------------------------
__global__ void node_init(const float* __restrict__ x, const float* __restrict__ gbias,
                          const float* __restrict__ niW1, const u16* __restrict__ w2t_g,
                          const float* __restrict__ b2, float* __restrict__ hf,
                          u16* __restrict__ hb, int Nn) {
  extern __shared__ char smem[];
  u16* sW2 = (u16*)smem;
  u16* sM  = (u16*)(smem + 32768);
  float* sw0 = (float*)(smem + 49152);
  float* sw1 = (float*)(smem + 49664);
  float* sgb = (float*)(smem + 50176);
  int tid = threadIdx.x;
#if USE_TDM
  if (tid == 0) {
    tdm_load_1d(w2t_g, sW2, 128 * 128);
    __builtin_amdgcn_s_wait_tensorcnt(0);
  }
#else
  {
    const uint4* g = (const uint4*)w2t_g;
    uint4* s = (uint4*)sW2;
    for (int i = tid; i < 2048; i += 128) s[i] = g[i];
  }
#endif
  sw0[tid] = niW1[tid]; sw1[tid] = niW1[128 + tid]; sgb[tid] = gbias[tid];
  int wave = tid >> 5, lane = tid & 31;
  int l15 = lane & 15, lhi = lane >> 4;
  int rowA = wave * 16 + l15;
  for (int t = 0; t < NTILES; ++t) {
    __syncthreads();
    int n0 = (blockIdx.x * NTILES + t) * 64;
    for (int idx = tid; idx < 64 * 128; idx += 128) {
      int nl = idx >> 7, jj = idx & 127;
      int node = n0 + nl; if (node >= Nn) node = Nn - 1;
      float v = x[2 * node] * sw0[jj] + x[2 * node + 1] * sw1[jj] + sgb[jj];
      sM[idx] = f2bf(silu_f(v));
    }
    __syncthreads();
    v8f c[8];
    zero_acc(c);
    gemm_acc<4, 128>(c, sM + rowA * 128, sW2, l15, lhi);
    float b2v[8];
#pragma unroll
    for (int n = 0; n < 8; ++n) b2v[n] = b2[n * 16 + l15];
#pragma unroll
    for (int v = 0; v < 8; ++v) {
      int node = n0 + wave * 16 + v + lhi * 8;
      if (node < Nn) {
#pragma unroll
        for (int n = 0; n < 8; ++n) {
          int col = n * 16 + l15;
          float val = c[n][v] + b2v[n];
          hf[(size_t)node * 128 + col] = val;
          hb[(size_t)node * 128 + col] = f2bf(val);
        }
      }
    }
  }
}

// ---------------------------------------------------------------------------
// Edge message MLP + scatter-add into agg.
// LDS: sW1[128][288] @0 (73728) | sW2[128][128] @73728 (32768)
//      sA[64][288] @106496 (36864) | sM[64][128] @143360 (16384)
//      sDst[64] @159744 (256)   -> 160000 B
// ---------------------------------------------------------------------------
__global__ void edge_msg(const u16* __restrict__ hbf, const u16* __restrict__ rbf,
                         const int* __restrict__ ei, const u16* __restrict__ w1t,
                         const u16* __restrict__ w2t, const float* __restrict__ b1,
                         const float* __restrict__ b2, float* __restrict__ agg) {
  extern __shared__ char smem[];
  u16* sW1 = (u16*)smem;
  u16* sW2 = (u16*)(smem + 73728);
  u16* sA  = (u16*)(smem + 106496);
  u16* sM  = (u16*)(smem + 143360);
  int* sDst = (int*)(smem + 159744);
  int tid = threadIdx.x;
#if USE_TDM
  if (tid == 0) {
    tdm_load_1d(w1t, sW1, 128 * 288);
    tdm_load_1d(w2t, sW2, 128 * 128);
    __builtin_amdgcn_s_wait_tensorcnt(0);
  }
#else
  {
    const uint4* g1 = (const uint4*)w1t; uint4* s1 = (uint4*)sW1;
    for (int i = tid; i < 4608; i += 128) s1[i] = g1[i];
    const uint4* g2 = (const uint4*)w2t; uint4* s2 = (uint4*)sW2;
    for (int i = tid; i < 2048; i += 128) s2[i] = g2[i];
  }
#endif
  int wave = tid >> 5, lane = tid & 31;
  int l15 = lane & 15, lhi = lane >> 4;
  int rowA = wave * 16 + l15;
  for (int t = 0; t < ETILES; ++t) {
    __syncthreads();
    int e0 = (blockIdx.x * ETILES + t) * 64;
    {
      int e = tid >> 1, sub = tid & 1;
      int ge = e0 + e;
      int srcI = ei[ge];
      int dstI = ei[N_EDGES + ge];
      if (sub == 0) sDst[e] = dstI;
      const uint4* hr = (const uint4*)(hbf + (size_t)(sub ? dstI : srcI) * 128);
      uint4* ar = (uint4*)(sA + e * 288 + sub * 128);
#pragma unroll
      for (int i = 0; i < 16; ++i) ar[i] = hr[i];
      if (sub) {
        const uint4* rr = (const uint4*)(rbf + (size_t)ge * 16);
        uint4* a2 = (uint4*)(sA + e * 288 + 256);
        a2[0] = rr[0]; a2[1] = rr[1];
        a2[2] = make_uint4(0, 0, 0, 0); a2[3] = make_uint4(0, 0, 0, 0);
      }
    }
    __syncthreads();
    v8f c[8];
    zero_acc(c);
    gemm_acc<9, 288>(c, sA + rowA * 288, sW1, l15, lhi);      // K = 288
    {
      float b1v[8];
#pragma unroll
      for (int n = 0; n < 8; ++n) b1v[n] = b1[n * 16 + l15];
#pragma unroll
      for (int n = 0; n < 8; ++n)
#pragma unroll
        for (int v = 0; v < 8; ++v)
          sM[(wave * 16 + v + lhi * 8) * 128 + n * 16 + l15] = f2bf(silu_f(c[n][v] + b1v[n]));
    }
    // per-wave rows only: LDS in-order within wave, no barrier needed
    zero_acc(c);
    gemm_acc<4, 128>(c, sM + rowA * 128, sW2, l15, lhi);      // K = 128
    {
      float b2v[8];
#pragma unroll
      for (int n = 0; n < 8; ++n) b2v[n] = b2[n * 16 + l15];
#pragma unroll
      for (int v = 0; v < 8; ++v) {
        int le = wave * 16 + v + lhi * 8;
        int dstI = sDst[le];
        float* arow = agg + (size_t)dstI * 128;
#pragma unroll
        for (int n = 0; n < 8; ++n)
          atomicAdd(arow + n * 16 + l15, c[n][v] + b2v[n]);
      }
    }
  }
}

// ---------------------------------------------------------------------------
// Node update: h += MLP([h | agg])     K = 256
// LDS: sW1[128][256] @0 (65536) | sW2 @65536 (32768) | sA[64][256] @98304 (32768)
//      sM[64][128] @131072 (16384)  -> 147456 B
// ---------------------------------------------------------------------------
__global__ void node_upd(const u16* __restrict__ hbf_in, const float* __restrict__ agg,
                         const u16* __restrict__ w1t, const u16* __restrict__ w2t,
                         const float* __restrict__ b1, const float* __restrict__ b2,
                         float* __restrict__ hf, u16* __restrict__ hb, int Nn) {
  extern __shared__ char smem[];
  u16* sW1 = (u16*)smem;
  u16* sW2 = (u16*)(smem + 65536);
  u16* sA  = (u16*)(smem + 98304);
  u16* sM  = (u16*)(smem + 131072);
  int tid = threadIdx.x;
#if USE_TDM
  if (tid == 0) {
    tdm_load_1d(w1t, sW1, 128 * 256);
    tdm_load_1d(w2t, sW2, 128 * 128);
    __builtin_amdgcn_s_wait_tensorcnt(0);
  }
#else
  {
    const uint4* g1 = (const uint4*)w1t; uint4* s1 = (uint4*)sW1;
    for (int i = tid; i < 4096; i += 128) s1[i] = g1[i];
    const uint4* g2 = (const uint4*)w2t; uint4* s2 = (uint4*)sW2;
    for (int i = tid; i < 2048; i += 128) s2[i] = g2[i];
  }
#endif
  int wave = tid >> 5, lane = tid & 31;
  int l15 = lane & 15, lhi = lane >> 4;
  int rowA = wave * 16 + l15;
  for (int t = 0; t < NTILES; ++t) {
    __syncthreads();
    int n0 = (blockIdx.x * NTILES + t) * 64;
    {
      int e = tid >> 1, sub = tid & 1;
      int node = n0 + e; if (node >= Nn) node = Nn - 1;
      if (sub == 0) {
        const uint4* hr = (const uint4*)(hbf_in + (size_t)node * 128);
        uint4* ar = (uint4*)(sA + e * 256);
#pragma unroll
        for (int i = 0; i < 16; ++i) ar[i] = hr[i];
      } else {
        const float4* ag = (const float4*)(agg + (size_t)node * 128);
        u16* dst = sA + e * 256 + 128;
#pragma unroll
        for (int i = 0; i < 32; ++i) {
          float4 f = ag[i];
          dst[4 * i + 0] = f2bf(f.x); dst[4 * i + 1] = f2bf(f.y);
          dst[4 * i + 2] = f2bf(f.z); dst[4 * i + 3] = f2bf(f.w);
        }
      }
    }
    __syncthreads();
    v8f c[8];
    zero_acc(c);
    gemm_acc<8, 256>(c, sA + rowA * 256, sW1, l15, lhi);      // K = 256
    {
      float b1v[8];
#pragma unroll
      for (int n = 0; n < 8; ++n) b1v[n] = b1[n * 16 + l15];
#pragma unroll
      for (int n = 0; n < 8; ++n)
#pragma unroll
        for (int v = 0; v < 8; ++v)
          sM[(wave * 16 + v + lhi * 8) * 128 + n * 16 + l15] = f2bf(silu_f(c[n][v] + b1v[n]));
    }
    zero_acc(c);
    gemm_acc<4, 128>(c, sM + rowA * 128, sW2, l15, lhi);      // K = 128
    {
      float b2v[8];
#pragma unroll
      for (int n = 0; n < 8; ++n) b2v[n] = b2[n * 16 + l15];
#pragma unroll
      for (int v = 0; v < 8; ++v) {
        int node = n0 + wave * 16 + v + lhi * 8;
        if (node < Nn) {
#pragma unroll
          for (int n = 0; n < 8; ++n) {
            int col = n * 16 + l15;
            float nv = hf[(size_t)node * 128 + col] + c[n][v] + b2v[n];
            hf[(size_t)node * 128 + col] = nv;
            hb[(size_t)node * 128 + col] = f2bf(nv);
          }
        }
      }
    }
  }
}

// ---------------------------------------------------------------------------
// Edge weight head: w = silu([h_src|h_dst|rbf] @ W1 + b1) @ W2 + b2
// then eps[dst] += (x[dst]-x[src]) * w   (atomics)
// LDS: sW1[128][288] @0 (73728) | sA[64][288] @73728 (36864)
//      sSrc @110592 (256) | sDst @110848 (256)  -> 111104 B
// ---------------------------------------------------------------------------
__global__ void wh_kernel(const u16* __restrict__ hbf, const u16* __restrict__ rbf,
                          const int* __restrict__ ei, const u16* __restrict__ w1t,
                          const float* __restrict__ b1, const float* __restrict__ w2,
                          const float* __restrict__ b2p, const float* __restrict__ x,
                          float* __restrict__ out) {
  extern __shared__ char smem[];
  u16* sW1 = (u16*)smem;
  u16* sA  = (u16*)(smem + 73728);
  int* sSrc = (int*)(smem + 110592);
  int* sDst = (int*)(smem + 110848);
  int tid = threadIdx.x;
#if USE_TDM
  if (tid == 0) {
    tdm_load_1d(w1t, sW1, 128 * 288);
    __builtin_amdgcn_s_wait_tensorcnt(0);
  }
#else
  {
    const uint4* g1 = (const uint4*)w1t; uint4* s1 = (uint4*)sW1;
    for (int i = tid; i < 4608; i += 128) s1[i] = g1[i];
  }
#endif
  int wave = tid >> 5, lane = tid & 31;
  int l15 = lane & 15, lhi = lane >> 4;
  int rowA = wave * 16 + l15;
  for (int t = 0; t < ETILES; ++t) {
    __syncthreads();
    int e0 = (blockIdx.x * ETILES + t) * 64;
    {
      int e = tid >> 1, sub = tid & 1;
      int ge = e0 + e;
      int srcI = ei[ge];
      int dstI = ei[N_EDGES + ge];
      if (sub == 0) { sSrc[e] = srcI; sDst[e] = dstI; }
      const uint4* hr = (const uint4*)(hbf + (size_t)(sub ? dstI : srcI) * 128);
      uint4* ar = (uint4*)(sA + e * 288 + sub * 128);
#pragma unroll
      for (int i = 0; i < 16; ++i) ar[i] = hr[i];
      if (sub) {
        const uint4* rr = (const uint4*)(rbf + (size_t)ge * 16);
        uint4* a2 = (uint4*)(sA + e * 288 + 256);
        a2[0] = rr[0]; a2[1] = rr[1];
        a2[2] = make_uint4(0, 0, 0, 0); a2[3] = make_uint4(0, 0, 0, 0);
      }
    }
    __syncthreads();
    v8f c[8];
    zero_acc(c);
    gemm_acc<9, 288>(c, sA + rowA * 288, sW1, l15, lhi);
    float b1v[8], w2v[8];
#pragma unroll
    for (int n = 0; n < 8; ++n) { b1v[n] = b1[n * 16 + l15]; w2v[n] = w2[n * 16 + l15]; }
    float whb2 = b2p[0];
#pragma unroll
    for (int v = 0; v < 8; ++v) {
      float p = 0.0f;
#pragma unroll
      for (int n = 0; n < 8; ++n) p += silu_f(c[n][v] + b1v[n]) * w2v[n];
      p += __shfl_xor(p, 1);
      p += __shfl_xor(p, 2);
      p += __shfl_xor(p, 4);
      p += __shfl_xor(p, 8);
      if (l15 == 0) {
        int le = wave * 16 + v + lhi * 8;
        float w = p + whb2;
        int s = sSrc[le], d = sDst[le];
        float cx = (x[2 * d] - x[2 * s]) * w;
        float cy = (x[2 * d + 1] - x[2 * s + 1]) * w;
        atomicAdd(out + 2 * d, cx);
        atomicAdd(out + 2 * d + 1, cy);
      }
    }
  }
}

// ---------------------------------------------------------------------------
extern "C" void kernel_launch(void* const* d_in, const int* in_sizes, int n_in,
                              void* d_out, int out_size, void* d_ws, size_t ws_size,
                              hipStream_t stream) {
  const float* x_t  = (const float*)d_in[0];
  const float* tval = (const float*)d_in[1];
  const float* cond = (const float*)d_in[2];
  const int*   ei   = (const int*)d_in[3];
  const float* tmW1 = (const float*)d_in[4];
  const float* tmb1 = (const float*)d_in[5];
  const float* tmW2 = (const float*)d_in[6];
  const float* tmb2 = (const float*)d_in[7];
  const float* cmW1 = (const float*)d_in[8];
  const float* cmb1 = (const float*)d_in[9];
  const float* cmW2 = (const float*)d_in[10];
  const float* cmb2 = (const float*)d_in[11];
  const float* niW1 = (const float*)d_in[12];
  const float* nib1 = (const float*)d_in[13];
  const float* niW2 = (const float*)d_in[14];
  const float* nib2 = (const float*)d_in[15];
  const float* msgW1 = (const float*)d_in[16];
  const float* msgb1 = (const float*)d_in[17];
  const float* msgW2 = (const float*)d_in[18];
  const float* msgb2 = (const float*)d_in[19];
  const float* updW1 = (const float*)d_in[20];
  const float* updb1 = (const float*)d_in[21];
  const float* updW2 = (const float*)d_in[22];
  const float* updb2 = (const float*)d_in[23];
  const float* whW1 = (const float*)d_in[24];
  const float* whb1 = (const float*)d_in[25];
  const float* whW2 = (const float*)d_in[26];
  const float* whb2 = (const float*)d_in[27];

  char* ws = (char*)d_ws;
  float* gbias   = (float*)(ws + 0);
  u16* niW2t     = (u16*)(ws + 512);
  u16* msgW1t    = (u16*)(ws + 33280);      // 4 x 128*288
  u16* msgW2t    = (u16*)(ws + 328192);     // 4 x 128*128
  u16* updW1t    = (u16*)(ws + 459264);     // 4 x 128*256
  u16* updW2t    = (u16*)(ws + 721408);     // 4 x 128*128
  u16* whW1t     = (u16*)(ws + 852480);     // 128*288
  float* hf      = (float*)(ws + 926208);   // N*128 f32
  u16* hb        = (u16*)(ws + 26526208);   // N*128 bf16
  float* agg     = (float*)(ws + 39326208); // N*128 f32
  u16* rbfbuf    = (u16*)(ws + 64926208);   // E*16 bf16

  hipMemsetAsync(d_out, 0, (size_t)N_NODES * 2 * sizeof(float), stream);

  prep_small<<<1, 128, 0, stream>>>(tval, cond, tmW1, tmb1, tmW2, tmb2,
                                    cmW1, cmb1, cmW2, cmb2, niW1, nib1, gbias);

  // weight prep (transpose + bf16)
  {
    int tot = 128 * 128;
    wt_kernel<<<(tot + 255) / 256, 256, 0, stream>>>(niW2, niW2t, 128, 128, 128);
    for (int l = 0; l < 4; ++l) {
      int t1 = 128 * 288;
      wt_kernel<<<(t1 + 255) / 256, 256, 0, stream>>>(msgW1 + (size_t)l * 272 * 128,
                                                      msgW1t + (size_t)l * 128 * 288, 272, 128, 288);
      wt_kernel<<<(tot + 255) / 256, 256, 0, stream>>>(msgW2 + (size_t)l * 128 * 128,
                                                       msgW2t + (size_t)l * 128 * 128, 128, 128, 128);
      int t2 = 128 * 256;
      wt_kernel<<<(t2 + 255) / 256, 256, 0, stream>>>(updW1 + (size_t)l * 256 * 128,
                                                      updW1t + (size_t)l * 128 * 256, 256, 128, 256);
      wt_kernel<<<(tot + 255) / 256, 256, 0, stream>>>(updW2 + (size_t)l * 128 * 128,
                                                       updW2t + (size_t)l * 128 * 128, 128, 128, 128);
    }
    int t1 = 128 * 288;
    wt_kernel<<<(t1 + 255) / 256, 256, 0, stream>>>(whW1, whW1t, 272, 128, 288);
  }

  rbf_kernel<<<(N_EDGES + 255) / 256, 256, 0, stream>>>(x_t, ei, rbfbuf);

  const int NB = (N_NODES + 64 * NTILES - 1) / (64 * NTILES);  // 196
  const int EB = N_EDGES / (64 * ETILES);                      // 3125

  node_init<<<NB, 128, 50688, stream>>>(x_t, gbias, niW1, niW2t, nib2, hf, hb, N_NODES);

  for (int l = 0; l < 4; ++l) {
    hipMemsetAsync(agg, 0, (size_t)N_NODES * 128 * sizeof(float), stream);
    edge_msg<<<EB, 128, 160000, stream>>>(hb, rbfbuf, ei,
                                          msgW1t + (size_t)l * 128 * 288,
                                          msgW2t + (size_t)l * 128 * 128,
                                          msgb1 + l * 128, msgb2 + l * 128, agg);
    node_upd<<<NB, 128, 147456, stream>>>(hb, agg,
                                          updW1t + (size_t)l * 128 * 256,
                                          updW2t + (size_t)l * 128 * 128,
                                          updb1 + l * 128, updb2 + l * 128,
                                          hf, hb, N_NODES);
  }

  wh_kernel<<<EB, 128, 111104, stream>>>(hb, rbfbuf, ei, whW1t, whb1, whW2, whb2,
                                         x_t, (float*)d_out);
}
